// Gemma4MoETextModelBackend_53815940219215
// MI455X (gfx1250) — matrix operand order repaired and checked by reading, not compiled, not run on hardware
//
#include <hip/hip_runtime.h>
#include <hip/hip_bf16.h>

typedef __attribute__((ext_vector_type(16))) _Float16 v16h;
typedef __attribute__((ext_vector_type(8)))  float    v8f;
typedef __attribute__((ext_vector_type(4)))  float    f4;
typedef __attribute__((ext_vector_type(4)))  int      i4;

#define H_DIM 2048
#define S_LEN 4096
#define NEXP  64
#define TOPK  4
#define SWIN  1024
#define EPSV  1e-6f
#define NEGV  (-3.4028234663852886e38f)   // jnp.finfo(float32).min

// ---------------------------------------------------------------------------
// Kernel 1: per-row sequential scan for vision-group ids (tiny: B*S ints)
// ---------------------------------------------------------------------------
__global__ void vision_gid_kernel(const int* __restrict__ mm, int* __restrict__ gid, int B) {
  int b = threadIdx.x;
  if (b >= B) return;
  const int* m = mm + (size_t)b * S_LEN;
  int* g = gid + (size_t)b * S_LEN;
  int cnt = 0, prev = 0;
  for (int s = 0; s < S_LEN; ++s) {
    int t = m[s];
    int isv = (t == 1 || t == 2) ? 1 : 0;
    cnt += (isv & (prev ^ 1));
    g[s] = isv ? (cnt - 1) : -1;
    prev = isv;
  }
}

// ---------------------------------------------------------------------------
// Kernel 2: fp32 -> f16 conversion of proj_w (256 KB, once per launch)
// ---------------------------------------------------------------------------
__global__ void cvt_f16_kernel(const float* __restrict__ w, _Float16* __restrict__ wh, int n) {
  int i = blockIdx.x * blockDim.x + threadIdx.x;
  if (i < n) wh[i] = (_Float16)w[i];
}

// ---------------------------------------------------------------------------
// Kernel 3: mask fill — store-bandwidth bound (268 MB), nontemporal f4 stores
// One block per (b,q) row; 256 threads x 4 kv elems x 4 iters = 4096 kv.
// ---------------------------------------------------------------------------
__global__ void __launch_bounds__(256)
mask_kernel(const int* __restrict__ packed, const int* __restrict__ gid,
            float* __restrict__ full_out, float* __restrict__ slide_out) {
  int row = blockIdx.x;            // b*S + q
  int b = row >> 12;               // /4096
  int q = row & (S_LEN - 1);
  int pq = packed[row];
  int gq = gid[row];
  const i4* pk4 = (const i4*)(packed + (size_t)b * S_LEN);
  const i4* gk4 = (const i4*)(gid    + (size_t)b * S_LEN);
  float* frow = full_out  + (size_t)row * S_LEN;
  float* srow = slide_out + (size_t)row * S_LEN;

  for (int i = threadIdx.x; i < S_LEN / 4; i += 256) {
    i4 pk = pk4[i];
    i4 gk = gk4[i];
    int kv0 = i * 4;
    f4 fo, so;
#pragma unroll
    for (int j = 0; j < 4; ++j) {
      int kv = kv0 + j;
      bool same_doc = (pq == pk[j]) && (pq > 0);
      bool causal   = (q >= kv);
      bool fullb    = same_doc && causal;
      bool samevg   = (gq == gk[j]) && (gq >= 0);
      bool slideb   = same_doc && ((causal && ((q - kv) < SWIN)) || samevg);
      fo[j] = fullb  ? 0.0f : NEGV;
      so[j] = slideb ? 0.0f : NEGV;
    }
    __builtin_nontemporal_store(fo, (f4*)(frow + kv0));
    __builtin_nontemporal_store(so, (f4*)(srow + kv0));
  }
}

// ---------------------------------------------------------------------------
// Kernel 4: MoE gate. 16 tokens / block, 128 threads (4 wave32).
//   Phase 1: per-wave RMSNorm of 4 tokens -> f16 xn in LDS
//   Phase 2: per-wave WMMA GEMM: C[16 tok x 16 experts], K=2048 in steps of 32
//   Phase 3: 16 lanes do softmax + top-4 + renorm over 64 experts
// ---------------------------------------------------------------------------
#define XN_STRIDE 2056   // halves; 4112 B rows: 16B-aligned, bank-rotating

__global__ void __launch_bounds__(128)
gate_kernel(const float* __restrict__ x, const float* __restrict__ scale,
            const _Float16* __restrict__ wh,
            float* __restrict__ weights, int* __restrict__ indices) {
  __shared__ __align__(16) _Float16 xn[16 * XN_STRIDE];
  __shared__ float logits[16 * NEXP];

  const int tid  = threadIdx.x;
  const int lane = tid & 31;
  const int wave = tid >> 5;
  const int blk  = blockIdx.x;
  const float invSqrtH = 0.022097086912079608f;  // 2048^-0.5

  // ---- Phase 1: RMSNorm(no scale) * H^-0.5 * scale -> f16 LDS -------------
  for (int j = 0; j < 4; ++j) {
    int t = wave * 4 + j;
    const float* xr = x + ((size_t)blk * 16 + t) * H_DIM;
    float vals[64];
    float ss = 0.f;
#pragma unroll
    for (int i = 0; i < 64; ++i) {
      float v = xr[lane + i * 32];
      vals[i] = v;
      ss += v * v;
    }
#pragma unroll
    for (int off = 16; off >= 1; off >>= 1)
      ss += __shfl_xor(ss, off, 32);
    float r = rsqrtf(ss * (1.0f / H_DIM) + EPSV) * invSqrtH;
    _Float16* rowp = xn + t * XN_STRIDE;
#pragma unroll
    for (int i = 0; i < 64; ++i) {
      int h = lane + i * 32;
      rowp[h] = (_Float16)(vals[i] * r * scale[h]);
    }
  }
  __syncthreads();

  // ---- Phase 2: WMMA GEMM; wave owns experts [wave*16, wave*16+16) --------
  union Frag16 { v16h v; uint4 q[2]; };
  v8f acc = {};
  const int m  = lane & 15;        // A row = token / B col = expert-in-tile
  const int kh = lane >> 4;        // K-half select
  const _Float16* arow  = xn + m * XN_STRIDE;
  const _Float16* bbase = wh + (size_t)(wave * 16 + m) * H_DIM;

  for (int ko = 0; ko < H_DIM; ko += 32) {
    Frag16 a, bf;
    // A 16x32 f16: lane holds K = kh*8 + {0..7} and 16 + kh*8 + {0..7}
    a.q[0] = *(const uint4*)(arow + ko + kh * 8);
    a.q[1] = *(const uint4*)(arow + ko + kh * 8 + 16);
    // B 32x16 f16: lane n holds K = kh*8 + {0..7} and 16 + kh*8 + {0..7}
    const _Float16* bp = bbase + ko + kh * 8;
    bf.q[0] = *(const uint4*)(bp);
    bf.q[1] = *(const uint4*)(bp + 16);
    acc = __builtin_amdgcn_wmma_f32_16x16x32_f16(
        /*neg_a=*/false, a.v, /*neg_b=*/false, bf.v,
        /*c_mod=*/(short)0, acc, /*reuse_a=*/false, /*reuse_b=*/false);
  }

  // C layout: VGPR r -> token M = r + 8*kh ; N (expert) = lane&15
#pragma unroll
  for (int r = 0; r < 8; ++r) {
    int tok = r + 8 * kh;
    logits[tok * NEXP + wave * 16 + m] = acc[r];
  }
  __syncthreads();

  // ---- Phase 3: softmax + top-4 + renormalize -----------------------------
  if (tid < 16) {
    const float* lr = logits + tid * NEXP;
    int selidx[TOPK];
    float selval[TOPK];
    unsigned long long taken = 0ull;
#pragma unroll
    for (int p = 0; p < TOPK; ++p) {
      float best = -__builtin_inff();
      int bi = 0;
      for (int e = 0; e < NEXP; ++e) {
        float v = lr[e];
        bool fre = !((taken >> e) & 1ull);
        if (fre && v > best) { best = v; bi = e; }   // stable: first index wins
      }
      taken |= (1ull << bi);
      selidx[p] = bi;
      selval[p] = best;
    }
    float mx = selval[0];                 // global max logit
    float ev[TOPK], ssum = 0.f;
#pragma unroll
    for (int p = 0; p < TOPK; ++p) { ev[p] = expf(selval[p] - mx); ssum += ev[p]; }
    float inv = 1.0f / ssum;              // softmax Z cancels in renorm
    size_t tok = (size_t)blk * 16 + tid;
#pragma unroll
    for (int p = 0; p < TOPK; ++p) {
      weights[tok * TOPK + p] = ev[p] * inv;
      indices[tok * TOPK + p] = selidx[p];
    }
  }
}

// ---------------------------------------------------------------------------
extern "C" void kernel_launch(void* const* d_in, const int* in_sizes, int n_in,
                              void* d_out, int out_size, void* d_ws, size_t ws_size,
                              hipStream_t stream) {
  const float* x      = (const float*)d_in[0];
  const int*   packed = (const int*)  d_in[1];
  const int*   mm     = (const int*)  d_in[2];
  const float* scale  = (const float*)d_in[3];
  const float* proj   = (const float*)d_in[4];

  const int BS = in_sizes[1];          // B * S
  const int B  = BS / S_LEN;
  const size_t SS = (size_t)S_LEN * S_LEN;

  float* full_out  = (float*)d_out;
  float* slide_out = full_out + (size_t)B * SS;
  float* weights   = slide_out + (size_t)B * SS;
  int*   indices   = (int*)(weights + (size_t)BS * TOPK);

  int* gid = (int*)d_ws;
  size_t gid_bytes = (((size_t)BS * sizeof(int)) + 255) & ~(size_t)255;
  _Float16* wh = (_Float16*)((char*)d_ws + gid_bytes);

  vision_gid_kernel<<<1, 32, 0, stream>>>(mm, gid, B);
  cvt_f16_kernel<<<(NEXP * H_DIM + 255) / 256, 256, 0, stream>>>(proj, wh, NEXP * H_DIM);
  mask_kernel<<<BS, 256, 0, stream>>>(packed, gid, full_out, slide_out);
  gate_kernel<<<BS / 16, 128, 0, stream>>>(x, scale, wh, weights, indices);
}
